// Enc_block_67920612819554
// MI455X (gfx1250) — matrix operand order repaired and checked by reading, not compiled, not run on hardware
//
#include <hip/hip_runtime.h>

// ---------------------------------------------------------------------------
// PointTransformerConv + down/pool layer for MI455X (gfx1250), bf16 WMMA path,
// with TDM (tensor_load_to_lds) weight staging in the per-edge kernel.
// ---------------------------------------------------------------------------

typedef __attribute__((ext_vector_type(16))) __bf16          v16bf;
typedef __attribute__((ext_vector_type(2)))  __bf16          bf16x2;
typedef __attribute__((ext_vector_type(8)))  float           v8f;
typedef __attribute__((ext_vector_type(16))) unsigned short  v16u;
typedef __attribute__((ext_vector_type(4)))  unsigned int    u32x4;
typedef __attribute__((ext_vector_type(8)))  int             i32x8;
typedef __attribute__((ext_vector_type(4)))  int             i32x4;

#define N_NODES 50000
#define N_EDGES 800000
#define CC      64
#define HH      128
#define N_POOL  25000
#define ENC_NEG_INF 0x007FFFFFu

#if __has_builtin(__builtin_amdgcn_tensor_load_to_lds)
#define HAVE_TDM 1
#endif

// ---------------- helpers ----------------

__device__ __forceinline__ unsigned short f2bf(float f) {
    // round-to-nearest-even f32 -> bf16 bits
    unsigned int u = __float_as_uint(f);
    unsigned int r = u + 0x7FFFu + ((u >> 16) & 1u);
    return (unsigned short)(r >> 16);
}
__device__ __forceinline__ __bf16 bfbits(unsigned short s) {
    return __builtin_bit_cast(__bf16, s);
}
__device__ __forceinline__ v16bf bcastv(v16u v) {
    return __builtin_bit_cast(v16bf, v);
}
// order-preserving f32 <-> u32 encoding for atomicMax-based segment max
__device__ __forceinline__ unsigned int fenc(float f) {
    unsigned int u = __float_as_uint(f);
    return (u & 0x80000000u) ? ~u : (u | 0x80000000u);
}
__device__ __forceinline__ float fdec(unsigned int u) {
    unsigned int v = (u & 0x80000000u) ? (u & 0x7FFFFFFFu) : ~u;
    return __uint_as_float(v);
}
__device__ __forceinline__ v8f vzero() {
    v8f z = {0.f, 0.f, 0.f, 0.f, 0.f, 0.f, 0.f, 0.f};
    return z;
}
__device__ __forceinline__ v8f wmma_bf16(v16bf a, v16bf b, v8f c) {
    // D = A(16x32 bf16) * B(32x16 bf16) + C(16x16 f32)
    return __builtin_amdgcn_wmma_f32_16x16x32_bf16(
        false, a, false, b, (short)0, c, false, false);
}
// Load one A-fragment (16x32 slice, k-step `ks`) from an LDS-staged bf16 tile
// with row stride 128 halfwords. Packed-pair (b32) loads.
__device__ __forceinline__ v16bf loadA_lds(const unsigned short* St, int lm, int h, int ks) {
    v16bf a;
#pragma unroll
    for (int j = 0; j < 8; ++j) {
        int kb = 2 * j + 8 * h + ((j >= 4) ? 8 : 0);
        unsigned int u = *(const unsigned int*)&St[lm * 128 + ks * 32 + kb];
        bf16x2 p = __builtin_bit_cast(bf16x2, u);
        a[2 * j]     = p[0];
        a[2 * j + 1] = p[1];
    }
    return a;
}

// ---------------- prep: swizzle weights into B-fragment lane order ----------

// dst[((ks*ntiles + t)*32 + lane)*16 + i] = bf16(W[k,n]); zero-pad k >= realK.
__global__ void swizzle_B_kernel(const float* __restrict__ W,
                                 unsigned short* __restrict__ dst,
                                 int Kp, int N, int realK) {
    int tid = blockIdx.x * blockDim.x + threadIdx.x;
    if (tid >= Kp * N) return;
    int i = tid & 15;
    int lane = (tid >> 4) & 31;
    int rest = tid >> 9;
    int ntiles = N >> 4;
    int t  = rest % ntiles;
    int ks = rest / ntiles;
    int h = lane >> 4, lm = lane & 15;
    int n = t * 16 + lm;
    int k = ks * 32 + i + 8 * h + ((i >= 8) ? 8 : 0);
    float v = (k < realK) ? W[k * N + n] : 0.0f;
    dst[tid] = f2bf(v);
}

// Swizzle row-major f32 [nrows x K] into A-fragment lane order (bf16).
__global__ void swizzle_A_kernel(const float* __restrict__ X,
                                 unsigned short* __restrict__ dst,
                                 int nrows, int K) {
    int tid = blockIdx.x * blockDim.x + threadIdx.x;
    if (tid >= nrows * K) return;
    int i = tid & 15;
    int lane = (tid >> 4) & 31;
    int rest = tid >> 9;
    int ksteps = K >> 5;
    int ks    = rest % ksteps;
    int rtile = rest / ksteps;
    int h = lane >> 4, lm = lane & 15;
    int row = rtile * 16 + lm;
    int k = ks * 32 + i + 8 * h + ((i >= 8) ? 8 : 0);
    dst[tid] = f2bf(X[(size_t)row * K + k]);
}

// ---------------- node-level GEMM: [16 rows] x [64x64] per wave -------------

__global__ __launch_bounds__(256) void gemm64_kernel(const v16u* __restrict__ A,
                                                     const v16u* __restrict__ B,
                                                     const float* __restrict__ bias,
                                                     float* __restrict__ out,
                                                     int nrtiles) {
    int lane  = threadIdx.x & 31;
    int rtile = (int)((blockIdx.x * blockDim.x + threadIdx.x) >> 5);
    if (rtile >= nrtiles) return;  // wave-uniform
    int h = lane >> 4, lm = lane & 15;
    v16bf a0 = bcastv(A[(rtile * 2 + 0) * 32 + lane]);
    v16bf a1 = bcastv(A[(rtile * 2 + 1) * 32 + lane]);
#pragma unroll
    for (int t = 0; t < 4; ++t) {
        v8f acc = vzero();
        acc = wmma_bf16(a0, bcastv(B[(0 * 4 + t) * 32 + lane]), acc);
        acc = wmma_bf16(a1, bcastv(B[(1 * 4 + t) * 32 + lane]), acc);
        int col = t * 16 + lm;
        float bv = bias ? bias[col] : 0.0f;
#pragma unroll
        for (int r = 0; r < 8; ++r) {
            int row = rtile * 16 + r + 8 * h;
            out[(size_t)row * 64 + col] = acc[r] + bv;
        }
    }
}

// ---------------- init ------------------------------------------------------

__global__ void init_kernel(unsigned int* __restrict__ amaxEnc,
                            float* __restrict__ denom,
                            float* __restrict__ outAcc,
                            unsigned int* __restrict__ pooledEnc, int n) {
    int i = blockIdx.x * blockDim.x + threadIdx.x;
    if (i < n) {
        amaxEnc[i]   = ENC_NEG_INF;
        denom[i]     = 0.f;
        outAcc[i]    = 0.f;
        pooledEnc[i] = ENC_NEG_INF;
    }
}

// ---------------- fused per-edge MLPs (pos MLP + attn MLP) ------------------
// One wave owns 16 edges; per-wave LDS traffic is DScnt-ordered (no barriers).
// Attention weights (32KB, contiguous B-fragment buffers) are staged to LDS
// once per block via the Tensor Data Mover.

#define WPB_E 4
#define ATT_W_U16 (64 * 128 + 128 * 64)   // 16384 u16 = 32KB

__global__ __launch_bounds__(32 * WPB_E) void edge_attn_kernel(
    const int* __restrict__ srcIdx, const int* __restrict__ dstIdx,
    const float* __restrict__ pos,
    const float* __restrict__ aSrcF, const float* __restrict__ aDstF,
    const unsigned short* __restrict__ Bpos1, const float* __restrict__ pos_b1,
    const unsigned short* __restrict__ Bpos2, const float* __restrict__ pos_b2,
    const unsigned short* __restrict__ BattGlob,   // wAtt1 ++ wAtt2, contiguous
    const float* __restrict__ attn_b1, const float* __restrict__ attn_b2,
    float* __restrict__ aBuf, float* __restrict__ deltaBuf,
    unsigned int* __restrict__ amaxEnc) {
    __shared__ unsigned short sStage[WPB_E][16 * 128];        // 16KB bf16 stage
    __shared__ float          sAdiff[WPB_E][16 * 64];         // 16KB adiff
    __shared__ __align__(32) unsigned short sW[ATT_W_U16];    // 32KB attn weights

    int lane  = threadIdx.x & 31;
    int w     = threadIdx.x >> 5;
    int wtile = blockIdx.x * WPB_E + w;    // grid is exact: E/16/WPB_E blocks
    int h = lane >> 4, lm = lane & 15;
    int e0 = wtile * 16;

    // ---- stage attention weights into LDS via TDM (wave 0 issues) ----
#if defined(HAVE_TDM)
    if (w == 0) {
        unsigned long long ga = (unsigned long long)BattGlob;
        // Generic LDS pointers carry the LDS byte offset in addr[31:0].
        unsigned int ldsOff = (unsigned int)(unsigned long long)(const void*)sW;
        u32x4 g0;
        g0[0] = 1u;                                       // count=1 (valid D#)
        g0[1] = ldsOff;                                   // lds_addr (bytes)
        g0[2] = (unsigned int)ga;                         // global_addr[31:0]
        g0[3] = (unsigned int)((ga >> 32) & 0x1FFFFFFull) // global_addr[56:32]
                | (2u << 30);                             // type=2 (image)
        i32x8 g1;
        g1[0] = 0x00010000;                // data_size=1 (2 bytes/element)
        g1[1] = (int)((ATT_W_U16 & 0xFFFF) << 16);        // tensor_dim0 lo16
        g1[2] = (int)((ATT_W_U16 >> 16) | (1 << 16));     // dim0 hi16 | dim1=1
        g1[3] = (int)((ATT_W_U16 & 0xFFFF) << 16);        // tile_dim0
        g1[4] = 0;                                        // tile_dim1/2 unused
        g1[5] = (int)ATT_W_U16;                           // dim0_stride lo32
        g1[6] = 0;
        g1[7] = 0;
        i32x4 gz  = {0, 0, 0, 0};
        i32x8 gz8 = {0, 0, 0, 0, 0, 0, 0, 0};
        // clang-23 / therock form: 6 args, extra int32x8 group before cpol.
        __builtin_amdgcn_tensor_load_to_lds(g0, g1, gz, gz, gz8, 0);
        __builtin_amdgcn_s_wait_tensorcnt(0);
    }
#else
    {
        const v16u* gsrc = (const v16u*)BattGlob;
        v16u* ldst = (v16u*)sW;
        for (int i = threadIdx.x; i < ATT_W_U16 / 16; i += blockDim.x)
            ldst[i] = gsrc[i];
    }
#endif
    __syncthreads();

    const v16u* B1 = (const v16u*)Bpos1;
    const v16u* B2 = (const v16u*)Bpos2;
    const v16u* B3 = (const v16u*)sW;                     // attn_w1 frags (LDS)
    const v16u* B4 = (const v16u*)(sW + 64 * 128);        // attn_w2 frags (LDS)
    unsigned short* St = sStage[w];
    float*          Ad = sAdiff[w];

    int eIdx = e0 + lm;
    int sI = srcIdx[eIdx], dI = dstIdx[eIdx];

    // Prefetch the randomly-gathered alpha rows (256B each) so the gather
    // latency hides behind the pos-MLP WMMA work below.
    __builtin_prefetch(aDstF + (size_t)dI * 64, 0, 3);
    __builtin_prefetch(aDstF + (size_t)dI * 64 + 32, 0, 3);
    __builtin_prefetch(aSrcF + (size_t)sI * 64, 0, 3);
    __builtin_prefetch(aSrcF + (size_t)sI * 64 + 32, 0, 3);

    float dp0 = pos[dI * 3 + 0] - pos[sI * 3 + 0];
    float dp1 = pos[dI * 3 + 1] - pos[sI * 3 + 1];
    float dp2 = pos[dI * 3 + 2] - pos[sI * 3 + 2];

    // A-frag for dpos, K padded 3 -> 32 (Bpos1 is zero-padded to match)
    v16bf aP;
#pragma unroll
    for (int i = 0; i < 16; ++i) {
        int k = i + 8 * h + ((i >= 8) ? 8 : 0);
        float v = (k == 0) ? dp0 : (k == 1) ? dp1 : (k == 2) ? dp2 : 0.0f;
        aP[i] = bfbits(f2bf(v));
    }

    // ---- pos MLP layer 1: h1 = relu(dpos @ pos_w1 + b1)  [16 x 128] ----
#pragma unroll
    for (int t = 0; t < 8; ++t) {
        v8f acc = wmma_bf16(aP, bcastv(B1[t * 32 + lane]), vzero());
        float bv = pos_b1[t * 16 + lm];
#pragma unroll
        for (int r = 0; r < 8; ++r) {
            float hv = fmaxf(acc[r] + bv, 0.0f);
            St[(r + 8 * h) * 128 + t * 16 + lm] = f2bf(hv);
        }
    }
    v16bf aH[4];
#pragma unroll
    for (int ks = 0; ks < 4; ++ks) aH[ks] = loadA_lds(St, lm, h, ks);

    // ---- pos MLP layer 2: delta = h1 @ pos_w2 + b2  [16 x 64] ----
    v8f accD[4];
#pragma unroll
    for (int t = 0; t < 4; ++t) {
        v8f acc = vzero();
#pragma unroll
        for (int ks = 0; ks < 4; ++ks)
            acc = wmma_bf16(aH[ks], bcastv(B2[(ks * 4 + t) * 32 + lane]), acc);
        float bv = pos_b2[t * 16 + lm];
#pragma unroll
        for (int r = 0; r < 8; ++r) {
            acc[r] += bv;
            int M = r + 8 * h;
            deltaBuf[(size_t)(e0 + M) * 64 + t * 16 + lm] = acc[r];
        }
        accD[t] = acc;
    }

    // ---- stage adiff = alpha_dst[dst] - alpha_src[src]  (coalesced) ----
    {
        const float2* AD = (const float2*)aDstF;
        const float2* AS = (const float2*)aSrcF;
#pragma unroll 4
        for (int m = 0; m < 16; ++m) {
            int sm = __shfl(sI, m, 32);
            int dm = __shfl(dI, m, 32);
            float2 d2 = AD[(size_t)dm * 32 + lane];
            float2 s2 = AS[(size_t)sm * 32 + lane];
            Ad[m * 64 + lane * 2 + 0] = d2.x - s2.x;
            Ad[m * 64 + lane * 2 + 1] = d2.y - s2.y;
        }
    }

    // ---- s = adiff + delta -> bf16 stage ----
#pragma unroll
    for (int t = 0; t < 4; ++t) {
#pragma unroll
        for (int r = 0; r < 8; ++r) {
            int M = r + 8 * h, col = t * 16 + lm;
            float sv = accD[t][r] + Ad[M * 64 + col];
            St[M * 128 + col] = f2bf(sv);
        }
    }
    v16bf aS[2];
#pragma unroll
    for (int ks = 0; ks < 2; ++ks) aS[ks] = loadA_lds(St, lm, h, ks);

    // ---- attn layer 1: a1 = relu(s @ attn_w1 + b1)  [16 x 128] ----
#pragma unroll
    for (int t = 0; t < 8; ++t) {
        v8f acc = vzero();
        acc = wmma_bf16(aS[0], bcastv(B3[(0 * 8 + t) * 32 + lane]), acc);
        acc = wmma_bf16(aS[1], bcastv(B3[(1 * 8 + t) * 32 + lane]), acc);
        float bv = attn_b1[t * 16 + lm];
#pragma unroll
        for (int r = 0; r < 8; ++r) {
            float hv = fmaxf(acc[r] + bv, 0.0f);
            St[(r + 8 * h) * 128 + t * 16 + lm] = f2bf(hv);
        }
    }
    v16bf aA[4];
#pragma unroll
    for (int ks = 0; ks < 4; ++ks) aA[ks] = loadA_lds(St, lm, h, ks);

    // ---- attn layer 2: a = a1 @ attn_w2 + b2; store + segment max ----
#pragma unroll
    for (int t = 0; t < 4; ++t) {
        v8f acc = vzero();
#pragma unroll
        for (int ks = 0; ks < 4; ++ks)
            acc = wmma_bf16(aA[ks], bcastv(B4[(ks * 4 + t) * 32 + lane]), acc);
        float bv = attn_b2[t * 16 + lm];
#pragma unroll
        for (int r = 0; r < 8; ++r) {
            int M = r + 8 * h, col = t * 16 + lm;
            float av = acc[r] + bv;
            aBuf[(size_t)(e0 + M) * 64 + col] = av;
            int dM = __shfl(dI, M, 32);
            atomicMax(&amaxEnc[(size_t)dM * 64 + col], fenc(av));
        }
    }
}

// ---------------- softmax passes & aggregation ------------------------------

__global__ void softmax_exp_kernel(float* __restrict__ aBuf,
                                   const int* __restrict__ dstIdx,
                                   const unsigned int* __restrict__ amaxEnc,
                                   float* __restrict__ denom, size_t total) {
    size_t i = (size_t)blockIdx.x * blockDim.x + threadIdx.x;
    if (i >= total) return;
    int e = (int)(i >> 6), c = (int)(i & 63);
    int d = dstIdx[e];
    float m  = fdec(amaxEnc[(size_t)d * 64 + c]);
    float ex = __expf(aBuf[i] - m);
    aBuf[i] = ex;
    atomicAdd(&denom[(size_t)d * 64 + c], ex);
}

__global__ void aggregate_kernel(const float* __restrict__ eBuf,
                                 const float* __restrict__ deltaBuf,
                                 const int* __restrict__ srcIdx,
                                 const int* __restrict__ dstIdx,
                                 const float* __restrict__ vF,
                                 const float* __restrict__ denom,
                                 float* __restrict__ outAcc, size_t total) {
    size_t i = (size_t)blockIdx.x * blockDim.x + threadIdx.x;
    if (i >= total) return;
    int e = (int)(i >> 6), c = (int)(i & 63);
    int s = srcIdx[e], d = dstIdx[e];
    float alpha = eBuf[i] / (denom[(size_t)d * 64 + c] + 1e-16f);
    float val   = alpha * (vF[(size_t)s * 64 + c] + deltaBuf[i]);
    atomicAdd(&outAcc[(size_t)d * 64 + c], val);
}

__global__ void pool_kernel(const float* __restrict__ hBuf,
                            const int* __restrict__ srcIdx,
                            const int* __restrict__ dstIdx,
                            unsigned int* __restrict__ pooledEnc, size_t total) {
    size_t i = (size_t)blockIdx.x * blockDim.x + threadIdx.x;
    if (i >= total) return;
    int e = (int)(i >> 6), c = (int)(i & 63);
    atomicMax(&pooledEnc[(size_t)dstIdx[e] * 64 + c],
              fenc(hBuf[(size_t)srcIdx[e] * 64 + c]));
}

__global__ void gather_kernel(const float* __restrict__ hBuf,
                              const unsigned int* __restrict__ pooledEnc,
                              const float* __restrict__ pos,
                              const int* __restrict__ poolIdx,
                              float* __restrict__ out) {
    int i = blockIdx.x * blockDim.x + threadIdx.x;
    const int nh = N_POOL * 64;
    if (i < nh) {
        int p = i >> 6, c = i & 63;
        int node = poolIdx[p];
        float hv = hBuf[(size_t)node * 64 + c];
        float pv = fdec(pooledEnc[(size_t)node * 64 + c]);
        out[i] = fmaxf(hv, pv);
    } else if (i < nh + N_POOL * 3) {
        int j = i - nh;
        int p = j / 3, r = j % 3;
        out[i] = pos[(size_t)poolIdx[p] * 3 + r];
    }
}

// ---------------- host launcher ---------------------------------------------

extern "C" void kernel_launch(void* const* d_in, const int* in_sizes, int n_in,
                              void* d_out, int out_size, void* d_ws, size_t ws_size,
                              hipStream_t stream) {
    (void)in_sizes; (void)n_in; (void)out_size; (void)ws_size;
    const float* x       = (const float*)d_in[0];
    const float* pos     = (const float*)d_in[1];
    const int*   eidx    = (const int*)d_in[2];
    const int*   poolIx  = (const int*)d_in[3];
    const float* lin_w   = (const float*)d_in[4];
    const float* lin_src = (const float*)d_in[5];
    const float* lin_dst = (const float*)d_in[6];
    const float* pos_w1  = (const float*)d_in[7];
    const float* pos_b1  = (const float*)d_in[8];
    const float* pos_w2  = (const float*)d_in[9];
    const float* pos_b2  = (const float*)d_in[10];
    const float* attn_w1 = (const float*)d_in[11];
    const float* attn_b1 = (const float*)d_in[12];
    const float* attn_w2 = (const float*)d_in[13];
    const float* attn_b2 = (const float*)d_in[14];
    const float* down_w  = (const float*)d_in[15];
    const float* down_b  = (const float*)d_in[16];
    const int* srcIdx = eidx;
    const int* dstIdx = eidx + N_EDGES;
    float* out = (float*)d_out;

    char* ws = (char*)d_ws;
    size_t off = 0;
    auto alloc = [&](size_t bytes) -> char* {
        char* p = ws + off;
        off = (off + bytes + 255) & ~(size_t)255;
        return p;
    };
    unsigned short* xA     = (unsigned short*)alloc((size_t)N_NODES * CC * 2);
    unsigned short* wLin   = (unsigned short*)alloc(64 * 64 * 2);
    unsigned short* wSrc   = (unsigned short*)alloc(64 * 64 * 2);
    unsigned short* wDst   = (unsigned short*)alloc(64 * 64 * 2);
    unsigned short* wPos1  = (unsigned short*)alloc(32 * 128 * 2);
    unsigned short* wPos2  = (unsigned short*)alloc(128 * 64 * 2);
    unsigned short* wAtt1  = (unsigned short*)alloc(64 * 128 * 2);   // contiguous
    unsigned short* wAtt2  = (unsigned short*)alloc(128 * 64 * 2);   // with wAtt1
    unsigned short* wDown  = (unsigned short*)alloc(64 * 64 * 2);
    float*          vF     = (float*)alloc((size_t)N_NODES * CC * 4);
    float*          aSrcF  = (float*)alloc((size_t)N_NODES * CC * 4);
    float*          aDstF  = (float*)alloc((size_t)N_NODES * CC * 4);
    float*          aBuf   = (float*)alloc((size_t)N_EDGES * CC * 4);
    float*          dltBuf = (float*)alloc((size_t)N_EDGES * CC * 4);
    unsigned int*   amaxE  = (unsigned int*)alloc((size_t)N_NODES * CC * 4);
    float*          denom  = (float*)alloc((size_t)N_NODES * CC * 4);
    float*          outAcc = (float*)alloc((size_t)N_NODES * CC * 4);
    unsigned short* outA   = (unsigned short*)alloc((size_t)N_NODES * CC * 2);
    float*          hBuf   = (float*)alloc((size_t)N_NODES * CC * 4);
    unsigned int*   poolE  = (unsigned int*)alloc((size_t)N_NODES * CC * 4);

    // 1) weight swizzles (B-fragment lane order, bf16; pos_w1 padded K 3->32)
    swizzle_B_kernel<<<(64 * 64 + 255) / 256, 256, 0, stream>>>(lin_w,   wLin,  64, 64, 64);
    swizzle_B_kernel<<<(64 * 64 + 255) / 256, 256, 0, stream>>>(lin_src, wSrc,  64, 64, 64);
    swizzle_B_kernel<<<(64 * 64 + 255) / 256, 256, 0, stream>>>(lin_dst, wDst,  64, 64, 64);
    swizzle_B_kernel<<<(32 * 128 + 255) / 256, 256, 0, stream>>>(pos_w1, wPos1, 32, 128, 3);
    swizzle_B_kernel<<<(128 * 64 + 255) / 256, 256, 0, stream>>>(pos_w2, wPos2, 128, 64, 128);
    swizzle_B_kernel<<<(64 * 128 + 255) / 256, 256, 0, stream>>>(attn_w1, wAtt1, 64, 128, 64);
    swizzle_B_kernel<<<(128 * 64 + 255) / 256, 256, 0, stream>>>(attn_w2, wAtt2, 128, 64, 128);
    swizzle_B_kernel<<<(64 * 64 + 255) / 256, 256, 0, stream>>>(down_w,  wDown, 64, 64, 64);

    // 2) x -> A-fragment order
    const int nodeElems = N_NODES * CC;
    swizzle_A_kernel<<<(nodeElems + 255) / 256, 256, 0, stream>>>(x, xA, N_NODES, CC);

    // 3) node linears: v, alpha_src, alpha_dst  (3125 row-tiles, 8 waves/block)
    const int nrt = N_NODES / 16;
    const int gemmBlocks = (nrt + 7) / 8;
    gemm64_kernel<<<gemmBlocks, 256, 0, stream>>>((const v16u*)xA, (const v16u*)wLin, nullptr, vF,    nrt);
    gemm64_kernel<<<gemmBlocks, 256, 0, stream>>>((const v16u*)xA, (const v16u*)wSrc, nullptr, aSrcF, nrt);
    gemm64_kernel<<<gemmBlocks, 256, 0, stream>>>((const v16u*)xA, (const v16u*)wDst, nullptr, aDstF, nrt);

    // 4) init accumulators
    init_kernel<<<(nodeElems + 255) / 256, 256, 0, stream>>>(amaxE, denom, outAcc, poolE, nodeElems);

    // 5) fused per-edge MLPs + segment max (exact grid: E/16/WPB_E blocks)
    edge_attn_kernel<<<N_EDGES / 16 / WPB_E, 32 * WPB_E, 0, stream>>>(
        srcIdx, dstIdx, pos, aSrcF, aDstF,
        wPos1, pos_b1, wPos2, pos_b2, wAtt1 /* ++ wAtt2 */, attn_b1, attn_b2,
        aBuf, dltBuf, amaxE);

    // 6) softmax numerator + denominator
    const size_t edgeElems = (size_t)N_EDGES * CC;
    const int edgeBlocks = (int)((edgeElems + 255) / 256);
    softmax_exp_kernel<<<edgeBlocks, 256, 0, stream>>>(aBuf, dstIdx, amaxE, denom, edgeElems);

    // 7) weighted aggregation
    aggregate_kernel<<<edgeBlocks, 256, 0, stream>>>(aBuf, dltBuf, srcIdx, dstIdx, vF, denom, outAcc, edgeElems);

    // 8) down layer: h = out @ down_w + down_b
    swizzle_A_kernel<<<(nodeElems + 255) / 256, 256, 0, stream>>>(outAcc, outA, N_NODES, CC);
    gemm64_kernel<<<gemmBlocks, 256, 0, stream>>>((const v16u*)outA, (const v16u*)wDown, down_b, hBuf, nrt);

    // 9) neighbor max pool
    pool_kernel<<<edgeBlocks, 256, 0, stream>>>(hBuf, srcIdx, dstIdx, poolE, edgeElems);

    // 10) final gather: max(h, pooled)[pool_index] ++ pos[pool_index]
    const int outElems = N_POOL * 64 + N_POOL * 3;
    gather_kernel<<<(outElems + 255) / 256, 256, 0, stream>>>(hBuf, poolE, pos, poolIx, out);
}
